// MultiHeadAttention_61881888801123
// MI455X (gfx1250) — matrix operand order repaired and checked by reading, not compiled, not run on hardware
//
#include <hip/hip_runtime.h>
#include <hip/hip_bf16.h>

// ---------------------------------------------------------------------------
// MultiHeadAttention on gfx1250 (CDNA5, wave32, WMMA 16x16x32 f16 -> f32 acc)
//
// Pipeline (all in d_ws, f16 staging):
//   1) cvt f32->f16: q,k,v, Wq,Wk,Wv,Wo
//   2) gemm_xw_t<0>: Qp = (q @ Wq^T) head-split  -> (B,H,S,64) f16
//      gemm_xw_t<0>: Kp = (k @ Wk^T) head-split  -> (B,H,S,64) f16
//      gemm_xw_t<1>: VpT= (v @ Wv^T) head-split, transposed -> (B,H,64,S) f16
//   3) flash_attn: online-softmax attention, ctx -> (B,S,1024) f16
//   4) gemm_xw_t<2>: out = ctx @ Wo^T -> f32 d_out
// ---------------------------------------------------------------------------

#define S_LEN  2048
#define DMODEL 1024
#define NHEADS 16
#define DHEAD  64
#define NBATCH 2
#define MTOT   (NBATCH * S_LEN)   // 4096
#define PT_LD  40                 // LDS P-tile row stride in halfs (80B = 5x16B)

typedef __attribute__((ext_vector_type(16))) _Float16     v16h;
typedef __attribute__((ext_vector_type(8)))  float        v8f;
typedef __attribute__((ext_vector_type(4)))  unsigned int v4u;

union F16Frag { v16h v; unsigned int u[8]; v4u q[2]; _Float16 h[16]; };

__device__ __forceinline__ v8f wmma_f16(v16h a, v16h b, v8f c) {
  // D = A(16x32) * B(32x16) + C(16x16 f32)
  return __builtin_amdgcn_wmma_f32_16x16x32_f16(
      /*neg_a=*/false, a, /*neg_b=*/false, b,
      /*c_mod=*/(short)0, c, /*reuse_a=*/false, /*reuse_b=*/false);
}

// A-matrix fragment 16x32 (MxK) f16. Element (m,k) read from base[m*ld + k].
// ISA layout: lane L holds row m = L&15; halfs 0..7 -> k = 8*(L>>4)+0..7,
// halfs 8..15 -> k = 8*(L>>4)+16..23.  Both runs contiguous -> two b128 loads.
// Requires ld % 8 == 0 and 16B-aligned base.
__device__ __forceinline__ v16h load_frag_a(const _Float16* __restrict__ base,
                                            int ld, int lane) {
  F16Frag f;
  const _Float16* p = base + (size_t)(lane & 15) * (size_t)ld + ((lane >> 4) << 3);
  f.q[0] = *(const v4u*)(p);       // k offsets +0 .. +7
  f.q[1] = *(const v4u*)(p + 16);  // k offsets +16 .. +23
  return f.v;
}

// B-matrix fragment 32x16 (KxN) f16, element (k,n) read from base[n*ld + k]
// (source is n-major with k contiguous -> A*B^T GEMM operands).
// ISA layout: lane L holds col n = L&15; halfs 0..7 -> k = 8*(L>>4)+0..7,
// halfs 8..15 -> k = 8*(L>>4)+16..23.
__device__ __forceinline__ v16h load_frag_b(const _Float16* __restrict__ base,
                                            int ld, int lane) {
  F16Frag f;
  const _Float16* p = base + (size_t)(lane & 15) * (size_t)ld + ((lane >> 4) << 3);
  f.q[0] = *(const v4u*)(p);
  f.q[1] = *(const v4u*)(p + 16);
  return f.v;
}

// ---------------------------------------------------------------------------
// f32 -> f16 conversion, 4 elements per thread
// ---------------------------------------------------------------------------
__global__ __launch_bounds__(256) void cvt_f32_to_f16(
    const float* __restrict__ src, _Float16* __restrict__ dst, int n4) {
  int i = blockIdx.x * blockDim.x + threadIdx.x;
  if (i < n4) {
    const float4 f = ((const float4*)src)[i];
    union { _Float16 h[4]; unsigned long long u; } o;
    o.h[0] = (_Float16)f.x; o.h[1] = (_Float16)f.y;
    o.h[2] = (_Float16)f.z; o.h[3] = (_Float16)f.w;
    ((unsigned long long*)dst)[i] = o.u;
  }
}

// ---------------------------------------------------------------------------
// C[M x 1024] = A[M x 1024] * W[1024 x 1024]^T, f16 in, WMMA f32 accumulate.
// Wave tile 64x64 (4x4 C-frags), workgroup 128 threads = 2x2 waves = 128x128.
// MODE 0: out f16 head-split (B,H,S,64)
// MODE 1: out f16 head-split transposed (B,H,64,S)
// MODE 2: out f32 plain (M,1024)
// ---------------------------------------------------------------------------
template <int MODE>
__global__ __launch_bounds__(128) void gemm_xw_t(const _Float16* __restrict__ A,
                                                 const _Float16* __restrict__ W,
                                                 void* __restrict__ Out) {
  const int lane  = threadIdx.x & 31;
  const int wv    = threadIdx.x >> 5;
  const int rbase = (lane >> 4) << 3;  // 8*(lane/16)
  const int cn    = lane & 15;
  const int m0 = blockIdx.y * 128 + (wv >> 1) * 64;
  const int n0 = blockIdx.x * 128 + (wv & 1) * 64;

  v8f c[4][4];
#pragma unroll
  for (int i = 0; i < 4; ++i)
#pragma unroll
    for (int j = 0; j < 4; ++j) c[i][j] = (v8f){0.f,0.f,0.f,0.f,0.f,0.f,0.f,0.f};

  for (int k0 = 0; k0 < DMODEL; k0 += 32) {
    if (k0 + 32 < DMODEL) {
      __builtin_prefetch(A + (size_t)m0 * DMODEL + k0 + 32, 0, 1);
      __builtin_prefetch(W + (size_t)n0 * DMODEL + k0 + 32, 0, 1);
    }
    v16h a[4], b[4];
#pragma unroll
    for (int i = 0; i < 4; ++i)
      a[i] = load_frag_a(A + (size_t)(m0 + 16 * i) * DMODEL + k0, DMODEL, lane);
#pragma unroll
    for (int j = 0; j < 4; ++j)
      b[j] = load_frag_b(W + (size_t)(n0 + 16 * j) * DMODEL + k0, DMODEL, lane);
#pragma unroll
    for (int i = 0; i < 4; ++i)
#pragma unroll
      for (int j = 0; j < 4; ++j) c[i][j] = wmma_f16(a[i], b[j], c[i][j]);
  }

#pragma unroll
  for (int i = 0; i < 4; ++i)
#pragma unroll
    for (int j = 0; j < 4; ++j)
#pragma unroll
      for (int r = 0; r < 8; ++r) {
        const int m = m0 + 16 * i + r + rbase;
        const int n = n0 + 16 * j + cn;
        const float val = c[i][j][r];
        if (MODE == 0) {
          const int bb = m >> 11, s = m & (S_LEN - 1);
          const int hh = n >> 6, d = n & (DHEAD - 1);
          ((_Float16*)Out)[((size_t)(bb * NHEADS + hh) * S_LEN + s) * DHEAD + d] =
              (_Float16)val;
        } else if (MODE == 1) {
          const int bb = m >> 11, s = m & (S_LEN - 1);
          const int hh = n >> 6, d = n & (DHEAD - 1);
          ((_Float16*)Out)[((size_t)(bb * NHEADS + hh) * DHEAD + d) * S_LEN + s] =
              (_Float16)val;
        } else {
          ((float*)Out)[(size_t)m * DMODEL + n] = val;
        }
      }
}

// ---------------------------------------------------------------------------
// Flash attention. Wave = 16 query rows, 4 waves/block = 64 rows/block.
// Grid: (S_LEN/64, NHEADS, NBATCH).
// S-frag row groups (lanes 0-15 / 16-31) match __shfl_xor(...,16) groups.
// ---------------------------------------------------------------------------
__global__ __launch_bounds__(128) void flash_attn_kernel(
    const _Float16* __restrict__ Qp, const _Float16* __restrict__ Kp,
    const _Float16* __restrict__ VpT, const unsigned char* __restrict__ mask,
    _Float16* __restrict__ ctx) {
  __shared__ _Float16 ptile[4][16][PT_LD];  // per-wave 16x32 P tile, 16B-aligned rows

  const int lane  = threadIdx.x & 31;
  const int wv    = threadIdx.x >> 5;
  const int rbase = (lane >> 4) << 3;
  const int cn    = lane & 15;
  const int b = blockIdx.z, h = blockIdx.y;
  const int m0 = blockIdx.x * 64 + wv * 16;  // query row block within sequence

  const _Float16* Qb = Qp  + (size_t)(b * NHEADS + h) * S_LEN * DHEAD;
  const _Float16* Kb = Kp  + (size_t)(b * NHEADS + h) * S_LEN * DHEAD;
  const _Float16* Vb = VpT + (size_t)(b * NHEADS + h) * DHEAD * S_LEN;
  const unsigned char* Mb = mask + (size_t)b * S_LEN * S_LEN;

  // Q fragments for k=0..31 and k=32..63, pre-scaled by 1/sqrt(64)=0.125 (exact in f16)
  F16Frag q0f, q1f;
  q0f.v = load_frag_a(Qb + (size_t)m0 * DHEAD, DHEAD, lane);
  q1f.v = load_frag_a(Qb + (size_t)m0 * DHEAD + 32, DHEAD, lane);
#pragma unroll
  for (int i = 0; i < 16; ++i) {
    q0f.h[i] = q0f.h[i] * (_Float16)0.125f;
    q1f.h[i] = q1f.h[i] * (_Float16)0.125f;
  }

  v8f o0 = (v8f){0.f,0.f,0.f,0.f,0.f,0.f,0.f,0.f};
  v8f o1 = o0, o2 = o0, o3 = o0;
  float mr[8], lr[8];
#pragma unroll
  for (int r = 0; r < 8; ++r) { mr[r] = -1e30f; lr[r] = 0.f; }

  _Float16* pt = &ptile[wv][0][0];

  for (int n0 = 0; n0 < S_LEN; n0 += 32) {
    // ---- S = (Q/8) * K^T for 32 key columns (two 16-col C frags) ----
    v8f s[2];
#pragma unroll
    for (int j = 0; j < 2; ++j) {
      const int nb = n0 + 16 * j;
      v16h kf0 = load_frag_b(Kb + (size_t)nb * DHEAD, DHEAD, lane);
      v16h kf1 = load_frag_b(Kb + (size_t)nb * DHEAD + 32, DHEAD, lane);
      v8f acc = (v8f){0.f,0.f,0.f,0.f,0.f,0.f,0.f,0.f};
      acc = wmma_f16(q0f.v, kf0, acc);
      acc = wmma_f16(q1f.v, kf1, acc);
      // mask: True -> -1e9 before scale == -1.25e8 after scale
      const unsigned char* mp = Mb + (size_t)(m0 + rbase) * S_LEN + nb + cn;
#pragma unroll
      for (int r = 0; r < 8; ++r)
        if (mp[r * S_LEN]) acc[r] = -1.25e8f;
      s[j] = acc;
    }

    // ---- online softmax update (per row r; row = 16-lane group) ----
    float alpha[8];
#pragma unroll
    for (int r = 0; r < 8; ++r) {
      float bm = fmaxf(s[0][r], s[1][r]);
#pragma unroll
      for (int i = 1; i < 16; i <<= 1) bm = fmaxf(bm, __shfl_xor(bm, i, 16));
      const float mn = fmaxf(mr[r], bm);
      alpha[r] = __expf(mr[r] - mn);
      mr[r] = mn;
      const float p0 = __expf(s[0][r] - mn);
      const float p1 = __expf(s[1][r] - mn);
      s[0][r] = p0; s[1][r] = p1;
      float bs = p0 + p1;
#pragma unroll
      for (int i = 1; i < 16; i <<= 1) bs += __shfl_xor(bs, i, 16);
      lr[r] = lr[r] * alpha[r] + bs;
    }

    // ---- C-layout P -> LDS -> A-layout P fragment ----
#pragma unroll
    for (int r = 0; r < 8; ++r) {
      pt[(r + rbase) * PT_LD + cn]      = (_Float16)s[0][r];
      pt[(r + rbase) * PT_LD + 16 + cn] = (_Float16)s[1][r];
    }
    __syncthreads();
    const v16h pf = load_frag_a(pt, PT_LD, lane);  // 2x ds_load_b128

    // ---- rescale running O, then O += P * V ----
#pragma unroll
    for (int r = 0; r < 8; ++r) {
      o0[r] *= alpha[r]; o1[r] *= alpha[r];
      o2[r] *= alpha[r]; o3[r] *= alpha[r];
    }
    const v16h vf0 = load_frag_b(Vb + (size_t)0  * 16 * S_LEN + n0, S_LEN, lane);
    const v16h vf1 = load_frag_b(Vb + (size_t)1  * 16 * S_LEN + n0, S_LEN, lane);
    const v16h vf2 = load_frag_b(Vb + (size_t)2  * 16 * S_LEN + n0, S_LEN, lane);
    const v16h vf3 = load_frag_b(Vb + (size_t)3  * 16 * S_LEN + n0, S_LEN, lane);
    o0 = wmma_f16(pf, vf0, o0);
    o1 = wmma_f16(pf, vf1, o1);
    o2 = wmma_f16(pf, vf2, o2);
    o3 = wmma_f16(pf, vf3, o3);
    __syncthreads();
  }

  // ---- normalize and write ctx (B, S, H*64) as f16 ----
#pragma unroll
  for (int r = 0; r < 8; ++r) {
    const size_t m = (size_t)b * S_LEN + m0 + r + rbase;
    _Float16* op = ctx + m * DMODEL + (size_t)h * DHEAD + cn;
    const float inv = 1.0f / lr[r];
    op[0]  = (_Float16)(o0[r] * inv);
    op[16] = (_Float16)(o1[r] * inv);
    op[32] = (_Float16)(o2[r] * inv);
    op[48] = (_Float16)(o3[r] * inv);
  }
}

// ---------------------------------------------------------------------------
// Host-side launch
// ---------------------------------------------------------------------------
extern "C" void kernel_launch(void* const* d_in, const int* in_sizes, int n_in,
                              void* d_out, int out_size, void* d_ws, size_t ws_size,
                              hipStream_t stream) {
  (void)in_sizes; (void)n_in; (void)out_size; (void)ws_size;
  const float* q  = (const float*)d_in[0];
  const float* k  = (const float*)d_in[1];
  const float* v  = (const float*)d_in[2];
  const unsigned char* mask = (const unsigned char*)d_in[3];  // jnp.bool_ = 1 byte
  const float* Wq = (const float*)d_in[4];
  const float* Wk = (const float*)d_in[5];
  const float* Wv = (const float*)d_in[6];
  const float* Wo = (const float*)d_in[7];

  constexpr size_t E_X = (size_t)MTOT * DMODEL;    // 4,194,304 halfs
  constexpr size_t E_W = (size_t)DMODEL * DMODEL;  // 1,048,576 halfs
  _Float16* ws   = (_Float16*)d_ws;                // total 64 MB of f16 staging
  _Float16* q16  = ws;
  _Float16* k16  = ws + E_X;
  _Float16* vh16 = ws + 2 * E_X;
  _Float16* Wq16 = ws + 3 * E_X;
  _Float16* Wk16 = Wq16 + E_W;
  _Float16* Wv16 = Wk16 + E_W;
  _Float16* Wo16 = Wv16 + E_W;
  _Float16* Qp   = Wo16 + E_W;   // (B,H,S,64)
  _Float16* Kp   = Qp + E_X;     // (B,H,S,64)
  _Float16* VpT  = Kp + E_X;     // (B,H,64,S)
  _Float16* ctx  = VpT + E_X;    // (B,S,1024)

  // 1) f32 -> f16 conversions
  {
    const int n4x = (int)(E_X / 4), n4w = (int)(E_W / 4);
    dim3 blk(256);
    cvt_f32_to_f16<<<dim3((n4x + 255) / 256), blk, 0, stream>>>(q,  q16,  n4x);
    cvt_f32_to_f16<<<dim3((n4x + 255) / 256), blk, 0, stream>>>(k,  k16,  n4x);
    cvt_f32_to_f16<<<dim3((n4x + 255) / 256), blk, 0, stream>>>(v,  vh16, n4x);
    cvt_f32_to_f16<<<dim3((n4w + 255) / 256), blk, 0, stream>>>(Wq, Wq16, n4w);
    cvt_f32_to_f16<<<dim3((n4w + 255) / 256), blk, 0, stream>>>(Wk, Wk16, n4w);
    cvt_f32_to_f16<<<dim3((n4w + 255) / 256), blk, 0, stream>>>(Wv, Wv16, n4w);
    cvt_f32_to_f16<<<dim3((n4w + 255) / 256), blk, 0, stream>>>(Wo, Wo16, n4w);
  }

  // 2) projections (WMMA GEMMs)
  const dim3 gg(DMODEL / 128, MTOT / 128), gb(128);
  gemm_xw_t<0><<<gg, gb, 0, stream>>>(q16,  Wq16, (void*)Qp);
  gemm_xw_t<0><<<gg, gb, 0, stream>>>(k16,  Wk16, (void*)Kp);
  gemm_xw_t<1><<<gg, gb, 0, stream>>>(vh16, Wv16, (void*)VpT);

  // 3) flash attention
  flash_attn_kernel<<<dim3(S_LEN / 64, NHEADS, NBATCH), dim3(128), 0, stream>>>(
      Qp, Kp, VpT, mask, ctx);

  // 4) output projection -> f32 d_out
  gemm_xw_t<2><<<gg, gb, 0, stream>>>(ctx, Wo16, d_out);
}
